// PointTransformerV3_34875134443596
// MI455X (gfx1250) — matrix operand order repaired and checked by reading, not compiled, not run on hardware
//
#include <hip/hip_runtime.h>
#include <hip/hip_bf16.h>
#include <cstdint>

// ---------------------------------------------------------------------------
// MI455X (gfx1250): bf16 WMMA 16x16x32 GEMMs with async global->LDS staging,
// windowed WMMA attention, deterministic bitonic Morton argsort.
// ---------------------------------------------------------------------------

typedef __attribute__((ext_vector_type(16))) __bf16 v16bf;
typedef __attribute__((ext_vector_type(8)))  __bf16 v8bf;
typedef __attribute__((ext_vector_type(8)))  float  v8f;

#define NPTS  196608
#define SORTN (1 << 18)

__device__ __forceinline__ float gelu_f(float x) {
  return 0.5f * x * (1.0f + erff(x * 0.70710678118654752f));
}

// Async DMA of one 16B chunk per lane into LDS (CDNA5 GLOBAL_LOAD_ASYNC_TO_LDS,
// tracked by ASYNCcnt; bypasses VGPRs).
__device__ __forceinline__ void async_cp16(uint32_t lds, const __bf16* g) {
  asm volatile("global_load_async_to_lds_b128 %0, %1, off"
               :: "v"(lds), "v"(g) : "memory");
}
__device__ __forceinline__ uint32_t lds_addr(const void* p) {
  return (uint32_t)(uintptr_t)p;   // low 32 bits of generic addr = LDS offset
}

// A-fragment (16x32 bf16): lane idx=lane&15 holds row idx; half=lane>>4 selects
// K interleave {half*8..+7, 16+half*8..+7} (ISA 7.12.2 16-bit A table).
__device__ __forceinline__ v16bf ldfragA(const __bf16* rowp, int half) {
  union { v16bf v; v8bf h[2]; } u;
  u.h[0] = *(const v8bf*)(rowp + half * 8);
  u.h[1] = *(const v8bf*)(rowp + half * 8 + 16);
  return u.v;
}
// B-fragment (32x16 bf16) from column-major storage: lane idx holds column idx,
// K interleave {half*8..+7, 16+half*8..+7} (same order as A fragment).
__device__ __forceinline__ v16bf ldfragB(const __bf16* colp, int half) {
  union { v16bf v; v8bf h[2]; } u;
  u.h[0] = *(const v8bf*)(colp + half * 8);
  u.h[1] = *(const v8bf*)(colp + half * 8 + 16);
  return u.v;
}

// ---------------------------------------------------------------------------
// Morton + bitonic sort (deterministic stable argsort surrogate)
// ---------------------------------------------------------------------------
__device__ __forceinline__ unsigned part1by2(unsigned x) {
  x &= 1023u;
  x = (x | (x << 16)) & 50331903u;
  x = (x | (x << 8))  & 50393103u;
  x = (x | (x << 4))  & 51130563u;
  x = (x | (x << 2))  & 153391689u;
  return x;
}

__global__ void morton_key_kernel(const float* __restrict__ coord,
                                  unsigned long long* __restrict__ keys, int n) {
  int r = blockIdx.x * blockDim.x + threadIdx.x;
  if (r >= SORTN) return;
  if (r < n) {
    int gx = (int)floorf(coord[r * 3 + 0] * 1024.0f);
    int gy = (int)floorf(coord[r * 3 + 1] * 1024.0f);
    int gz = (int)floorf(coord[r * 3 + 2] * 1024.0f);
    gx = gx < 0 ? 0 : (gx > 1023 ? 1023 : gx);
    gy = gy < 0 ? 0 : (gy > 1023 ? 1023 : gy);
    gz = gz < 0 ? 0 : (gz > 1023 ? 1023 : gz);
    unsigned code = part1by2((unsigned)gx) | (part1by2((unsigned)gy) << 1) |
                    (part1by2((unsigned)gz) << 2);
    keys[r] = ((unsigned long long)code << 32) | (unsigned)r;
  } else {
    keys[r] = 0xFFFFFFFFFFFFFFFFull;
  }
}

__global__ void bitonic_step_kernel(unsigned long long* __restrict__ a, int j, int k) {
  int i = blockIdx.x * blockDim.x + threadIdx.x;
  int ixj = i ^ j;
  if (ixj > i) {
    unsigned long long x = a[i], y = a[ixj];
    bool up = ((i & k) == 0);
    if ((x > y) == up) { a[i] = y; a[ixj] = x; }
  }
}

__global__ void gather6_kernel(const float* __restrict__ feat,
                               const unsigned long long* __restrict__ keys,
                               float* __restrict__ xg, int n) {
  int i = blockIdx.x * blockDim.x + threadIdx.x;
  if (i >= n * 6) return;
  int r = i / 6, c = i % 6;
  unsigned ord = (unsigned)(keys[r] & 0xFFFFFFFFull);
  xg[i] = feat[(size_t)ord * 6 + c];
}

// ---------------------------------------------------------------------------
// Weight prep: fp32 [K x M] -> bf16 transposed [Mpad x K], zero padded rows.
// Runs once per launch; makes GEMM staging a pure 16B-chunk async copy.
// ---------------------------------------------------------------------------
__global__ void transposeW_kernel(const float* __restrict__ w, __bf16* __restrict__ wt,
                                  int K, int M, int Mp) {
  int i = blockIdx.x * blockDim.x + threadIdx.x;
  if (i >= Mp * K) return;
  int m = i / K, k = i % K;
  wt[i] = (m < M) ? (__bf16)w[(size_t)k * M + m] : (__bf16)0.0f;
}

// ---------------------------------------------------------------------------
// Small scalar kernels: embed (K=6), head (M=2), convert, pool, combine
// ---------------------------------------------------------------------------
__global__ void embed_kernel(const float* __restrict__ xg, const float* __restrict__ w,
                             const float* __restrict__ b, float* __restrict__ y, int n) {
  int i = blockIdx.x * blockDim.x + threadIdx.x;
  if (i >= n * 32) return;
  int r = i >> 5, c = i & 31;
  float a = b[c];
#pragma unroll
  for (int k = 0; k < 6; k++) a += xg[(size_t)r * 6 + k] * w[k * 32 + c];
  y[i] = a;
}

__global__ void head2_kernel(const float* __restrict__ x, const float* __restrict__ w2,
                             const float* __restrict__ b2, float* __restrict__ out, int n) {
  int r = blockIdx.x * blockDim.x + threadIdx.x;
  if (r >= n) return;
  float a0 = b2[0], a1 = b2[1];
#pragma unroll
  for (int c = 0; c < 32; c++) {
    float v = x[(size_t)r * 32 + c];
    a0 += v * w2[c * 2 + 0];
    a1 += v * w2[c * 2 + 1];
  }
  out[(size_t)r * 2 + 0] = a0;
  out[(size_t)r * 2 + 1] = a1;
}

__global__ void f2b_kernel(const float* __restrict__ in, __bf16* __restrict__ out, int n) {
  int i = blockIdx.x * blockDim.x + threadIdx.x;
  if (i < n) out[i] = (__bf16)in[i];
}

__global__ void maxpool2_kernel(const float* __restrict__ in, float* __restrict__ out,
                                int rowsOut, int C) {
  int i = blockIdx.x * blockDim.x + threadIdx.x;
  if (i >= rowsOut * C) return;
  int r = i / C, c = i % C;
  out[i] = fmaxf(in[(size_t)(2 * r) * C + c], in[(size_t)(2 * r + 1) * C + c]);
}

__global__ void combine_kernel(const float* __restrict__ sk, const float* __restrict__ par,
                               float* __restrict__ out, int rows, int C) {
  int i = blockIdx.x * blockDim.x + threadIdx.x;
  if (i >= rows * C) return;
  int r = i / C, c = i % C;
  out[i] = sk[i] + par[(size_t)(r >> 1) * C + c];
}

// ---------------------------------------------------------------------------
// LayerNorm (fp32 in -> bf16 out), one wave per row
// ---------------------------------------------------------------------------
__global__ __launch_bounds__(256) void ln_kernel(const float* __restrict__ x,
                                                 __bf16* __restrict__ out,
                                                 const float* __restrict__ g,
                                                 const float* __restrict__ b, int C) {
  int wave = threadIdx.x >> 5, lane = threadIdx.x & 31;
  int row = blockIdx.x * 8 + wave;
  const float* xr = x + (size_t)row * C;
  float s = 0.0f, s2 = 0.0f;
  for (int c = lane; c < C; c += 32) { float v = xr[c]; s += v; s2 += v * v; }
  for (int m = 16; m >= 1; m >>= 1) {
    s  += __shfl_xor(s,  m, 32);
    s2 += __shfl_xor(s2, m, 32);
  }
  float mean = s / C;
  float var  = s2 / C - mean * mean;
  float inv  = rsqrtf(var + 1e-5f);
  __bf16* orow = out + (size_t)row * C;
  for (int c = lane; c < C; c += 32)
    orow[c] = (__bf16)((xr[c] - mean) * inv * g[c] + b[c]);
}

// ---------------------------------------------------------------------------
// BatchNorm (global stats over rows) + exact GELU, in place
// ---------------------------------------------------------------------------
__global__ void bn_stats_kernel(const float* __restrict__ x, float* __restrict__ accum,
                                int rows, int C, int rpb) {
  int c = threadIdx.x;                    // blockDim.x == C
  int r0 = blockIdx.x * rpb;
  int r1 = r0 + rpb; if (r1 > rows) r1 = rows;
  float s = 0.0f, s2 = 0.0f;
  for (int r = r0; r < r1; r++) {
    float v = x[(size_t)r * C + c];
    s += v; s2 += v * v;
  }
  atomicAdd(&accum[c], s);
  atomicAdd(&accum[C + c], s2);
}

__global__ void bn_apply_gelu_kernel(const float* __restrict__ x, float* __restrict__ y,
                                     const float* __restrict__ g, const float* __restrict__ b,
                                     const float* __restrict__ accum, int rows, int C, int total) {
  int i = blockIdx.x * blockDim.x + threadIdx.x;
  if (i >= total) return;
  int c = i % C;
  float mean = accum[c] / rows;
  float var  = accum[C + c] / rows - mean * mean;
  float v = (x[i] - mean) * rsqrtf(var + 1e-5f) * g[c] + b[c];
  y[i] = gelu_f(v);
}

// ---------------------------------------------------------------------------
// bf16 GEMM: out[N x M] = A[N x K](bf16) @ W + bias, with WT = bf16 [Mpad x K]
// (pre-transposed, zero-padded). 64x64 tile / 256-thread WG; 8 waves each own a
// 16x32 strip (2 WMMAs/k-step). Double-buffered LDS, async global->LDS staging
// pipelined one k-step ahead; last k-step peeled so the hot loop is branch-free.
// EPI: 0 fp32, 1 fp32+residual(in place), 2 bf16, 3 bf16+GELU.
// Requires N%64==0, K%32==0.
// ---------------------------------------------------------------------------
template <int EPI>
__global__ __launch_bounds__(256) void gemm_bf16_kernel(
    const __bf16* __restrict__ A, const __bf16* __restrict__ WT,
    const float* __restrict__ bias, const float* __restrict__ res,
    float* __restrict__ outF, __bf16* __restrict__ outB, int K, int M) {
  __shared__ __bf16 As[2][64 * 40];   // 64x32 tiles, stride 40 (conflict-free)
  __shared__ __bf16 Ws[2][64 * 40];

  const int tid  = threadIdx.x;
  const int wave = tid >> 5, lane = tid & 31;
  const int idx  = lane & 15, half = lane >> 4;
  const int n0 = blockIdx.x * 64;
  const int m0 = blockIdx.y * 64;
  const int wr = wave & 3, wc = wave >> 2;

  // staging coords: thread t owns 16B chunk (row rr, cols cc..cc+7)
  const int rr = tid >> 2, cc = (tid & 3) * 8;
  const __bf16* gA = A  + (size_t)(n0 + rr) * K + cc;
  const __bf16* gW = WT + (size_t)(m0 + rr) * K + cc;   // Mpad rows: no guard
  uint32_t lA[2], lW[2];
  lA[0] = lds_addr(&As[0][rr * 40 + cc]);
  lA[1] = lds_addr(&As[1][rr * 40 + cc]);
  lW[0] = lds_addr(&Ws[0][rr * 40 + cc]);
  lW[1] = lds_addr(&Ws[1][rr * 40 + cc]);

  // prologue: stage k-step 0 into buffer 0
  async_cp16(lA[0], gA);
  async_cp16(lW[0], gW);

  v8f acc0 = {}; v8f acc1 = {};
  const int nsteps = K >> 5;
  int cur = 0;
  // steady state: branch-free (issue next -> wait old -> barrier -> WMMA -> barrier)
  for (int it = 0; it < nsteps - 1; it++) {
    const int nxt = cur ^ 1;
    async_cp16(lA[nxt], gA + ((size_t)(it + 1) << 5));
    async_cp16(lW[nxt], gW + ((size_t)(it + 1) << 5));
    asm volatile("s_wait_asynccnt 2" ::: "memory");   // old 2 retired
    __syncthreads();   // all waves' DMA for `cur` landed
    v16bf a  = ldfragA(&As[cur][(wr * 16 + idx) * 40], half);
    v16bf b0 = ldfragB(&Ws[cur][(wc * 32 + idx) * 40], half);
    v16bf b1 = ldfragB(&Ws[cur][(wc * 32 + 16 + idx) * 40], half);
    acc0 = __builtin_amdgcn_wmma_f32_16x16x32_bf16(false, a, false, b0, (short)0, acc0, false, false);
    acc1 = __builtin_amdgcn_wmma_f32_16x16x32_bf16(false, a, false, b1, (short)0, acc1, false, false);
    __syncthreads();   // `cur` free for the DMA issued next iteration
    cur = nxt;
  }
  // peeled final k-step
  asm volatile("s_wait_asynccnt 0" ::: "memory");
  __syncthreads();
  {
    v16bf a  = ldfragA(&As[cur][(wr * 16 + idx) * 40], half);
    v16bf b0 = ldfragB(&Ws[cur][(wc * 32 + idx) * 40], half);
    v16bf b1 = ldfragB(&Ws[cur][(wc * 32 + 16 + idx) * 40], half);
    acc0 = __builtin_amdgcn_wmma_f32_16x16x32_bf16(false, a, false, b0, (short)0, acc0, false, false);
    acc1 = __builtin_amdgcn_wmma_f32_16x16x32_bf16(false, a, false, b1, (short)0, acc1, false, false);
  }

  const int rbase = n0 + wr * 16 + half * 8;
  const int c0 = m0 + wc * 32 + idx;
#pragma unroll
  for (int v = 0; v < 8; v++) {
    const int r = rbase + v;
#pragma unroll
    for (int t = 0; t < 2; t++) {
      const int c = c0 + t * 16;
      const float av = (t == 0) ? acc0[v] : acc1[v];
      if (c < M) {
        float val = av + bias[c];
        size_t o = (size_t)r * M + c;
        if (EPI == 0)      outF[o] = val;
        else if (EPI == 1) outF[o] = res[o] + val;
        else if (EPI == 2) outB[o] = (__bf16)val;
        else               outB[o] = (__bf16)gelu_f(val);
      }
    }
  }
}

// ---------------------------------------------------------------------------
// Windowed attention, PATCH=48, one wave per (window, head). D = head dim.
// S = Q K^T (9 WMMAs, K padded to 32), softmax via lane-half shuffles in the
// native C/D layout, P -> LDS (overlays Q/K), O = P V (2 k-steps of WMMA).
// Q/K staged with async global->LDS; V transposed via ds stores.
// ---------------------------------------------------------------------------
template <int D>
__global__ __launch_bounds__(128) void attn_kernel(const __bf16* __restrict__ qkv,
                                                   __bf16* __restrict__ o,
                                                   int C, int heads) {
  __shared__ __bf16 smem[4 * 6144];  // 48 KB: 4 waves x 12288 B
  const int wave = threadIdx.x >> 5, lane = threadIdx.x & 31;
  const int idx = lane & 15, half = lane >> 4;
  __bf16* Qs = smem + wave * 6144;   // 48 x 40
  __bf16* Ks = Qs + 1920;            // 48 x 40
  __bf16* Vt = Qs + 3840;            // 32 x 72 (V transposed, K padded to 64)
  __bf16* Ps = Qs;                   // 48 x 72 overlay (P, cols padded to 64)

  const int task = blockIdx.x * 4 + wave;
  const int win = task / heads, h = task % heads;
  const size_t rowBase = (size_t)win * 48;
  const int str = 3 * C;
  const __bf16* qp = qkv + rowBase * str + h * D;
  const __bf16* kp = qp + C;
  const __bf16* vp = qp + 2 * C;

  // stage Q/K rows via async DMA (pad K-dim to 32 for D==16 with ds stores)
  for (int r = lane; r < 48; r += 32) {
#pragma unroll
    for (int j = 0; j < D; j += 8) {
      async_cp16(lds_addr(&Qs[r * 40 + j]), &qp[(size_t)r * str + j]);
      async_cp16(lds_addr(&Ks[r * 40 + j]), &kp[(size_t)r * str + j]);
    }
    if (D == 16) {
      v8bf z = {};
      *(v8bf*)&Qs[r * 40 + 16] = z; *(v8bf*)&Qs[r * 40 + 24] = z;
      *(v8bf*)&Ks[r * 40 + 16] = z; *(v8bf*)&Ks[r * 40 + 24] = z;
    }
  }
  // V transposed: Vt[j][r] = V[r][j], pad k=48..63 with zeros
  for (int t = lane; t < 48 * D; t += 32) {
    int r = t / D, j = t % D;
    Vt[j * 72 + r] = vp[(size_t)r * str + j];
  }
  for (int t = lane; t < D * 16; t += 32) {
    int j = t >> 4, k = 48 + (t & 15);
    Vt[j * 72 + k] = (__bf16)0.0f;
  }
  asm volatile("s_wait_asynccnt 0" ::: "memory");
  __syncthreads();

  v16bf aQ[3], bK[3];
#pragma unroll
  for (int tr = 0; tr < 3; tr++) aQ[tr] = ldfragA(&Qs[(tr * 16 + idx) * 40], half);
#pragma unroll
  for (int tc = 0; tc < 3; tc++) bK[tc] = ldfragB(&Ks[(tc * 16 + idx) * 40], half);

  v8f acc[3][3];
#pragma unroll
  for (int tr = 0; tr < 3; tr++)
#pragma unroll
    for (int tc = 0; tc < 3; tc++) {
      v8f z = {};
      acc[tr][tc] = __builtin_amdgcn_wmma_f32_16x16x32_bf16(
          false, aQ[tr], false, bK[tc], (short)0, z, false, false);
    }

  const float scale = (D == 16) ? 0.25f : 0.17677669529663689f;  // 1/sqrt(D)
  // softmax over the 48 columns (3 tiles x 16 lanes of one half) + write P
#pragma unroll
  for (int tr = 0; tr < 3; tr++) {
#pragma unroll
    for (int v = 0; v < 8; v++) {
      float x0 = acc[tr][0][v] * scale;
      float x1 = acc[tr][1][v] * scale;
      float x2 = acc[tr][2][v] * scale;
      float m = fmaxf(x0, fmaxf(x1, x2));
      for (int mk = 1; mk <= 8; mk <<= 1) m = fmaxf(m, __shfl_xor(m, mk, 32));
      float e0 = expf(x0 - m), e1 = expf(x1 - m), e2 = expf(x2 - m);
      float s = e0 + e1 + e2;
      for (int mk = 1; mk <= 8; mk <<= 1) s += __shfl_xor(s, mk, 32);
      float inv = 1.0f / s;
      int row = tr * 16 + half * 8 + v;
      Ps[row * 72 +  0 + idx] = (__bf16)(e0 * inv);
      Ps[row * 72 + 16 + idx] = (__bf16)(e1 * inv);
      Ps[row * 72 + 32 + idx] = (__bf16)(e2 * inv);
      Ps[row * 72 + 48 + idx] = (__bf16)0.0f;  // zero pad cols 48..63
    }
  }
  __syncthreads();

  // O = P @ V : hoist V fragments (invariant over row tiles)
  v16bf bV0[D / 16], bV1[D / 16];
#pragma unroll
  for (int tc = 0; tc < D / 16; tc++) {
    bV0[tc] = ldfragB(&Vt[(tc * 16 + idx) * 72 + 0],  half);
    bV1[tc] = ldfragB(&Vt[(tc * 16 + idx) * 72 + 32], half);
  }
#pragma unroll
  for (int tr = 0; tr < 3; tr++) {
    v16bf a0 = ldfragA(&Ps[(tr * 16 + idx) * 72 + 0],  half);
    v16bf a1 = ldfragA(&Ps[(tr * 16 + idx) * 72 + 32], half);
#pragma unroll
    for (int tc = 0; tc < D / 16; tc++) {
      v8f z = {};
      v8f t = __builtin_amdgcn_wmma_f32_16x16x32_bf16(false, a0, false, bV0[tc], (short)0, z, false, false);
      t = __builtin_amdgcn_wmma_f32_16x16x32_bf16(false, a1, false, bV1[tc], (short)0, t, false, false);
#pragma unroll
      for (int v = 0; v < 8; v++) {
        int row = tr * 16 + half * 8 + v;
        int col = tc * 16 + idx;
        o[(rowBase + row) * C + h * D + col] = (__bf16)t[v];
      }
    }
  }
}

// ---------------------------------------------------------------------------
// Host orchestration
// ---------------------------------------------------------------------------
struct BlockP {
  const float *fc1_b, *fc1_w, *fc2_b, *fc2_w, *ln1_b, *ln1_g, *ln2_b, *ln2_g,
              *proj_b, *proj_w, *qkv_b, *qkv_w;
};
struct BlockW { const __bf16 *qkv, *proj, *fc1, *fc2; };
struct Bufs { __bf16 *hB, *qkv, *oB, *hid; float* accum; };

static BlockP read_block(void* const* din, int& i) {
  BlockP p;
  p.fc1_b = (const float*)din[i++];  p.fc1_w = (const float*)din[i++];
  p.fc2_b = (const float*)din[i++];  p.fc2_w = (const float*)din[i++];
  p.ln1_b = (const float*)din[i++];  p.ln1_g = (const float*)din[i++];
  p.ln2_b = (const float*)din[i++];  p.ln2_g = (const float*)din[i++];
  p.proj_b = (const float*)din[i++]; p.proj_w = (const float*)din[i++];
  p.qkv_b = (const float*)din[i++];  p.qkv_w = (const float*)din[i++];
  return p;
}

template <int EPI>
static void run_gemm(const __bf16* A, const __bf16* WT, const float* bias,
                     const float* res, float* outF, __bf16* outB,
                     int Nr, int K, int M, hipStream_t s) {
  dim3 g(Nr / 64, (M + 63) / 64);
  gemm_bf16_kernel<EPI><<<g, 256, 0, s>>>(A, WT, bias, res, outF, outB, K, M);
}

static void run_ln(const float* x, __bf16* out, const float* g, const float* b,
                   int Nr, int C, hipStream_t s) {
  ln_kernel<<<Nr / 8, 256, 0, s>>>(x, out, g, b, C);
}

static void run_bn_gelu(float* x, const float* g, const float* b, int Nr, int C,
                        float* accum, hipStream_t s) {
  hipMemsetAsync(accum, 0, 2 * (size_t)C * sizeof(float), s);
  const int rpb = 2048;
  bn_stats_kernel<<<(Nr + rpb - 1) / rpb, C, 0, s>>>(x, accum, Nr, C, rpb);
  int total = Nr * C;
  bn_apply_gelu_kernel<<<(total + 255) / 256, 256, 0, s>>>(x, x, g, b, accum, Nr, C, total);
}

static void run_block(float* x, int Nr, int C, int heads, const BlockP& p,
                      const BlockW& w, const Bufs& B, hipStream_t s) {
  run_ln(x, B.hB, p.ln1_g, p.ln1_b, Nr, C, s);
  run_gemm<2>(B.hB, w.qkv, p.qkv_b, nullptr, nullptr, B.qkv, Nr, C, 3 * C, s);
  int D = C / heads;
  int blocks = (Nr / 48) * heads / 4;
  if (D == 16) attn_kernel<16><<<blocks, 128, 0, s>>>(B.qkv, B.oB, C, heads);
  else         attn_kernel<32><<<blocks, 128, 0, s>>>(B.qkv, B.oB, C, heads);
  run_gemm<1>(B.oB, w.proj, p.proj_b, x, x, nullptr, Nr, C, C, s);
  run_ln(x, B.hB, p.ln2_g, p.ln2_b, Nr, C, s);
  run_gemm<3>(B.hB, w.fc1, p.fc1_b, nullptr, nullptr, B.hid, Nr, C, 4 * C, s);
  run_gemm<1>(B.hid, w.fc2, p.fc2_b, x, x, nullptr, Nr, 4 * C, C, s);
}

extern "C" void kernel_launch(void* const* d_in, const int* in_sizes, int n_in,
                              void* d_out, int out_size, void* d_ws, size_t ws_size,
                              hipStream_t stream) {
  (void)in_sizes; (void)n_in; (void)out_size; (void)ws_size;
  const int N = NPTS;
  const int CH[4] = {32, 64, 128, 256};
  const int HEADS[4] = {2, 4, 8, 8};

  const float* coord = (const float*)d_in[0];
  const float* feat  = (const float*)d_in[1];

  // Params flattened in jax pytree order (dict keys sorted alphabetically):
  // dec, embed, enc, head, pool, unpool
  int i = 2;
  BlockP dec[3][2];
  for (int s0 = 0; s0 < 3; s0++)
    for (int b0 = 0; b0 < 2; b0++) dec[s0][b0] = read_block(d_in, i);
  const float* e_b  = (const float*)d_in[i++];
  const float* e_bb = (const float*)d_in[i++];
  const float* e_g  = (const float*)d_in[i++];
  const float* e_w  = (const float*)d_in[i++];
  BlockP enc[4][2];
  for (int s0 = 0; s0 < 4; s0++)
    for (int b0 = 0; b0 < 2; b0++) enc[s0][b0] = read_block(d_in, i);
  const float* h_b1 = (const float*)d_in[i++];
  const float* h_b2 = (const float*)d_in[i++];
  const float* h_bb = (const float*)d_in[i++];
  const float* h_g  = (const float*)d_in[i++];
  const float* h_w1 = (const float*)d_in[i++];
  const float* h_w2 = (const float*)d_in[i++];
  struct { const float *b, *bb, *g, *w; } pool[3];
  for (int k = 0; k < 3; k++) {
    pool[k].b  = (const float*)d_in[i++]; pool[k].bb = (const float*)d_in[i++];
    pool[k].g  = (const float*)d_in[i++]; pool[k].w  = (const float*)d_in[i++];
  }
  struct { const float *pb, *pbb, *pg, *pw, *sb, *sbb, *sg, *sw; } up[3];
  for (int k = 0; k < 3; k++) {
    up[k].pb = (const float*)d_in[i++]; up[k].pbb = (const float*)d_in[i++];
    up[k].pg = (const float*)d_in[i++]; up[k].pw  = (const float*)d_in[i++];
    up[k].sb = (const float*)d_in[i++]; up[k].sbb = (const float*)d_in[i++];
    up[k].sg = (const float*)d_in[i++]; up[k].sw  = (const float*)d_in[i++];
  }

  // workspace bump allocator
  char* base = (char*)d_ws;
  size_t off = 0;
  auto alloc = [&](size_t bytes) -> void* {
    void* r = base + off;
    off = (off + bytes + 255) & ~(size_t)255;
    return r;
  };
  const size_t EA = (size_t)N * 32;  // elems of every stage activation (equal)
  unsigned long long* keys = (unsigned long long*)alloc((size_t)SORTN * 8);
  float* xg = (float*)alloc((size_t)N * 6 * 4);
  float* x0 = (float*)alloc(EA * 4);   // stage0 x (also skip0)
  float* x1 = (float*)alloc(EA * 4);   // stage1 (skip1)
  float* x2 = (float*)alloc(EA * 4);   // stage2 (skip2)
  float* x3 = (float*)alloc(EA * 4);   // stage3
  float* dx[3];
  dx[0] = (float*)alloc(EA * 4);       // decoder C=128
  dx[1] = (float*)alloc(EA * 4);       // decoder C=64
  dx[2] = (float*)alloc(EA * 4);       // decoder C=32
  Bufs B;
  B.hB   = (__bf16*)alloc(EA * 2);
  B.qkv  = (__bf16*)alloc(EA * 3 * 2);
  B.oB   = (__bf16*)alloc(EA * 2);
  B.hid  = (__bf16*)alloc(EA * 4 * 2);
  float* tmp   = (float*)alloc(EA * 2 * 4);
  float* tmp2  = (float*)alloc(EA * 4);
  float* accum = (float*)alloc(512 * 4);

  // one-time weight prep: bf16 transposed, rows padded to multiple of 64
  auto prepW = [&](const float* w, int K, int M) -> const __bf16* {
    int Mp = (M + 63) & ~63;
    __bf16* wt = (__bf16*)alloc((size_t)Mp * K * 2);
    int total = Mp * K;
    transposeW_kernel<<<(total + 255) / 256, 256, 0, stream>>>(w, wt, K, M, Mp);
    return wt;
  };
  auto prepBlock = [&](const BlockP& p, int C) -> BlockW {
    BlockW w;
    w.qkv  = prepW(p.qkv_w,  C,     3 * C);
    w.proj = prepW(p.proj_w, C,     C);
    w.fc1  = prepW(p.fc1_w,  C,     4 * C);
    w.fc2  = prepW(p.fc2_w,  4 * C, C);
    return w;
  };
  BlockW encW[4][2], decW[3][2];
  for (int s0 = 0; s0 < 4; s0++)
    for (int b0 = 0; b0 < 2; b0++) encW[s0][b0] = prepBlock(enc[s0][b0], CH[s0]);
  for (int j = 0; j < 3; j++)
    for (int b0 = 0; b0 < 2; b0++) decW[j][b0] = prepBlock(dec[j][b0], CH[2 - j]);
  const __bf16* poolW[3];
  for (int k = 0; k < 3; k++) poolW[k] = prepW(pool[k].w, CH[k], CH[k + 1]);
  const __bf16 *upPW[3], *upSW[3];
  for (int j = 0; j < 3; j++) {
    int i2 = 2 - j;
    upPW[j] = prepW(up[j].pw, CH[i2 + 1], CH[i2]);
    upSW[j] = prepW(up[j].sw, CH[i2], CH[i2]);
  }
  const __bf16* headW1 = prepW(h_w1, 32, 32);

  // ---- Morton order (deterministic bitonic sort of (code<<32)|idx) ----
  morton_key_kernel<<<SORTN / 256, 256, 0, stream>>>(coord, keys, N);
  for (int k = 2; k <= SORTN; k <<= 1)
    for (int j = k >> 1; j > 0; j >>= 1)
      bitonic_step_kernel<<<SORTN / 256, 256, 0, stream>>>(keys, j, k);
  gather6_kernel<<<(N * 6 + 255) / 256, 256, 0, stream>>>(feat, keys, xg, N);

  // ---- Embed: x = gelu(bn(xg @ w + b)) ----
  embed_kernel<<<(N * 32 + 255) / 256, 256, 0, stream>>>(xg, e_w, e_b, x0, N);
  run_bn_gelu(x0, e_g, e_bb, N, 32, accum, stream);

  // ---- Encoder ----
  int Ns[4] = {N, N / 2, N / 4, N / 8};
  float* xs[4] = {x0, x1, x2, x3};
  for (int s0 = 0; s0 < 4; s0++) {
    for (int b0 = 0; b0 < 2; b0++)
      run_block(xs[s0], Ns[s0], CH[s0], HEADS[s0], enc[s0][b0], encW[s0][b0], B, stream);
    if (s0 < 3) {
      int C = CH[s0], C1 = CH[s0 + 1], Nr = Ns[s0];
      f2b_kernel<<<(Nr * C + 255) / 256, 256, 0, stream>>>(xs[s0], B.hB, Nr * C);
      run_gemm<0>(B.hB, poolW[s0], pool[s0].b, nullptr, tmp, nullptr, Nr, C, C1, stream);
      maxpool2_kernel<<<((Nr / 2) * C1 + 255) / 256, 256, 0, stream>>>(tmp, xs[s0 + 1], Nr / 2, C1);
      run_bn_gelu(xs[s0 + 1], pool[s0].g, pool[s0].bb, Nr / 2, C1, accum, stream);
    }
  }

  // ---- Decoder ----
  float* cur = x3;
  int curRows = Ns[3], curC = CH[3];
  float* skips[3] = {x0, x1, x2};
  for (int j = 0; j < 3; j++) {
    int i2 = 2 - j;
    int Cf = CH[i2];
    int skRows = Ns[i2];
    // par = gelu(bn(cur @ pw + pb))
    f2b_kernel<<<(curRows * curC + 255) / 256, 256, 0, stream>>>(cur, B.hB, curRows * curC);
    run_gemm<0>(B.hB, upPW[j], up[j].pb, nullptr, tmp, nullptr, curRows, curC, Cf, stream);
    run_bn_gelu(tmp, up[j].pg, up[j].pbb, curRows, Cf, accum, stream);
    // sk = gelu(bn(skip @ sw + sb))
    f2b_kernel<<<(skRows * Cf + 255) / 256, 256, 0, stream>>>(skips[i2], B.hB, skRows * Cf);
    run_gemm<0>(B.hB, upSW[j], up[j].sb, nullptr, tmp2, nullptr, skRows, Cf, Cf, stream);
    run_bn_gelu(tmp2, up[j].sg, up[j].sbb, skRows, Cf, accum, stream);
    // x = sk + repeat(par, 2)
    combine_kernel<<<(skRows * Cf + 255) / 256, 256, 0, stream>>>(tmp2, tmp, dx[j], skRows, Cf);
    cur = dx[j]; curRows = skRows; curC = Cf;
    for (int b0 = 0; b0 < 2; b0++)
      run_block(cur, curRows, Cf, HEADS[i2], dec[j][b0], decW[j][b0], B, stream);
  }

  // ---- Head ----
  f2b_kernel<<<(N * 32 + 255) / 256, 256, 0, stream>>>(cur, B.hB, N * 32);
  run_gemm<0>(B.hB, headW1, h_b1, nullptr, tmp, nullptr, N, 32, 32, stream);
  run_bn_gelu(tmp, h_g, h_bb, N, 32, accum, stream);
  head2_kernel<<<(N + 255) / 256, 256, 0, stream>>>(tmp, h_w2, h_b2, (float*)d_out, N);
}
